// DefDCTFFN_81535659147931
// MI455X (gfx1250) — matrix operand order repaired and checked by reading, not compiled, not run on hardware
//
#include <hip/hip_runtime.h>
#include <math.h>

#define HW   36864
#define Hd   192
#define Wd   192
#define HI   194          // padded spatial (pad=1)

typedef __bf16 bf16_t;
typedef __attribute__((ext_vector_type(16))) __bf16 v16bf;
typedef __attribute__((ext_vector_type(8)))  float  v8f;

__device__ inline bf16_t f2bf(float f) {
  unsigned u = __builtin_bit_cast(unsigned, f);
  u += 0x7FFFu + ((u >> 16) & 1u);            // round-to-nearest-even
  unsigned short s = (unsigned short)(u >> 16);
  return __builtin_bit_cast(bf16_t, s);
}

__device__ inline float gelu_f(float x) {
  return 0.5f * x * (1.0f + erff(x * 0.70710678118654752f));
}

// ---- gfx1250 async global->LDS staging (ASYNCcnt path), with safe fallback ----
#if defined(__gfx1250__) && __has_builtin(__builtin_amdgcn_global_load_async_to_lds_b128)
#define HAS_ASYNC_LDS 1
#endif

typedef int v4i __attribute__((vector_size(16)));
typedef __attribute__((address_space(1))) v4i* g_v4i_p;
typedef __attribute__((address_space(3))) v4i* l_v4i_p;

__device__ inline void cp16_g2l(const void* g, void* l) {
#if HAS_ASYNC_LDS
  __builtin_amdgcn_global_load_async_to_lds_b128(
      (g_v4i_p)const_cast<void*>(g), (l_v4i_p)l, 0, 0);
#else
  *(float4*)l = *(const float4*)g;
#endif
}

__device__ inline void cp_g2l_wait() {
#if HAS_ASYNC_LDS
#if __has_builtin(__builtin_amdgcn_s_wait_asynccnt)
  __builtin_amdgcn_s_wait_asynccnt(0);
#endif
#endif
}

enum { EPI_STORE = 0, EPI_PAD = 1, EPI_BIAS = 2, EPI_GATE = 3 };

// C[M][HW] = W[M x K] @ X[K][HW]; W,X are bf16 in HBM, async-staged to LDS,
// f32 accumulate via v_wmma_f32_16x16x32_bf16. 256 threads = 8 waves,
// 64-pixel N-slab per block, whole W LDS-resident.
template <int M, int K, int EPI>
__global__ __launch_bounds__(256) void gemm_wmma(
    const bf16_t* __restrict__ Wm, const bf16_t* __restrict__ X,
    void* __restrict__ Cm, const float* __restrict__ bias,
    const float* __restrict__ aux)
{
  __shared__ alignas(16) bf16_t sW[M * K];
  __shared__ alignas(16) bf16_t sX[K * 64];
  const int tid = threadIdx.x;
  const int p0  = blockIdx.x * 64;

  // async stage W: M*K bf16 in 16B chunks
  for (int c = tid; c < (M * K) / 8; c += 256)
    cp16_g2l(Wm + (size_t)c * 8, sW + (size_t)c * 8);
  // async stage X tile: row k contributes 64 bf16 = 8 chunks
  for (int c = tid; c < K * 8; c += 256) {
    int k = c >> 3, n = (c & 7) * 8;
    cp16_g2l(X + (size_t)k * HW + p0 + n, sX + k * 64 + n);
  }
  cp_g2l_wait();
  __syncthreads();

  const int wave  = tid >> 5;
  const int lane  = tid & 31;
  const int half  = lane >> 4;
  const int l16   = lane & 15;
  const int ntile = wave & 3;

  for (int mt = (wave >> 2); mt < M / 16; mt += 2) {
    v8f acc = {0.f, 0.f, 0.f, 0.f, 0.f, 0.f, 0.f, 0.f};
#pragma unroll
    for (int ks = 0; ks < K / 32; ++ks) {
      v16bf Af, Bf;
      const int arow = mt * 16 + l16;
      const int col  = ntile * 16 + l16;
#pragma unroll
      for (int j = 0; j < 8; ++j) {
        // A 16x32 bf16: lanes0-15 K {0..7,16..23}, lanes16-31 K {8..15,24..31}
        int ka = ((j & 3) << 1) + (half << 3) + ((j >> 2) << 4) + ks * 32;
        Af[2 * j]     = sW[arow * K + ka];
        Af[2 * j + 1] = sW[arow * K + ka + 1];
        // B 32x16 bf16: lanes0-15 K {0..7,16..23}, lanes16-31 K {8..15,24..31} (2 per VGPR)
        int kb = ((j & 3) << 1) + (half << 3) + ((j >> 2) << 4) + ks * 32;
        Bf[2 * j]     = sX[kb * 64 + col];
        Bf[2 * j + 1] = sX[(kb + 1) * 64 + col];
      }
      acc = __builtin_amdgcn_wmma_f32_16x16x32_bf16(false, Af, false, Bf,
                                                    (short)0, acc, false, false);
    }
    const int gcol = p0 + ntile * 16 + l16;
#pragma unroll
    for (int r = 0; r < 8; ++r) {
      int row = mt * 16 + r + (half << 3);     // D layout: M = vgpr + 8*half
      float v = acc[r];
      if (EPI == EPI_PAD || EPI == EPI_BIAS || EPI == EPI_GATE) v += bias[row];
      if (EPI == EPI_PAD) {
        int hh = gcol / Wd, ww = gcol - hh * Wd;
        ((float*)Cm)[(size_t)row * (HI * HI) + (size_t)(hh + 1) * HI + (ww + 1)] = v;
      } else if (EPI == EPI_GATE) {
        ((bf16_t*)Cm)[(size_t)row * HW + gcol] =
            f2bf(v * aux[(size_t)row * HW + gcol]);
      } else {
        ((float*)Cm)[(size_t)row * HW + gcol] = v;
      }
    }
  }
}

// in-place per-channel 8x8 DCT -> *mix -> IDCT; thread = channel, block = 8x8 tile.
// Channels 128..255 additionally emit a bf16 mirror (input_proj GEMM operand).
__global__ __launch_bounds__(256) void dct_k(float* __restrict__ y,
                                             const float* __restrict__ mix,
                                             bf16_t* __restrict__ x2b)
{
  __shared__ float sA[64];
  const int t = threadIdx.x;
  if (t < 64) {
    int k = t >> 3, i = t & 7;
    float c = (k == 0) ? 0.35355339059327373f : 0.5f;
    sA[t] = c * cosf(3.14159265358979323f * (2.f * i + 1.f) * (float)k / 16.f);
  }
  __syncthreads();
  const int by = blockIdx.x / 24, bx = blockIdx.x % 24;
  const size_t boff = (size_t)(by * 8) * Wd + bx * 8;
  float* base = y + (size_t)t * HW + boff;
  float Y[64], T[64];
#pragma unroll
  for (int r = 0; r < 8; ++r)
#pragma unroll
    for (int c = 0; c < 8; ++c) Y[r * 8 + c] = base[r * Wd + c];
  // T = A @ Y
#pragma unroll
  for (int i = 0; i < 8; ++i)
#pragma unroll
    for (int c = 0; c < 8; ++c) {
      float s = 0.f;
#pragma unroll
      for (int r = 0; r < 8; ++r) s += sA[i * 8 + r] * Y[r * 8 + c];
      T[i * 8 + c] = s;
    }
  // Y = (T @ A^T) .* mix
  const float* mx = mix + (size_t)t * 64;
#pragma unroll
  for (int i = 0; i < 8; ++i)
#pragma unroll
    for (int j = 0; j < 8; ++j) {
      float s = 0.f;
#pragma unroll
      for (int c = 0; c < 8; ++c) s += T[i * 8 + c] * sA[j * 8 + c];
      Y[i * 8 + j] = s * mx[i * 8 + j];
    }
  // T = A^T @ Y
#pragma unroll
  for (int r = 0; r < 8; ++r)
#pragma unroll
    for (int j = 0; j < 8; ++j) {
      float s = 0.f;
#pragma unroll
      for (int i = 0; i < 8; ++i) s += sA[i * 8 + r] * Y[i * 8 + j];
      T[r * 8 + j] = s;
    }
  // out = T @ A  (+ bf16 mirror for x2 half)
  bf16_t* mb = x2b + (size_t)(t - 128) * HW + boff;
#pragma unroll
  for (int r = 0; r < 8; ++r)
#pragma unroll
    for (int c = 0; c < 8; ++c) {
      float s = 0.f;
#pragma unroll
      for (int j = 0; j < 8; ++j) s += T[r * 8 + j] * sA[j * 8 + c];
      base[r * Wd + c] = s;
      if (t >= 128) mb[r * Wd + c] = f2bf(s);
    }
}

// depthwise 3x3 (pad 1) + exact GELU, channel-major (gate branch, f32 aux)
__global__ __launch_bounds__(256) void dw_gelu_k(const float* __restrict__ xin,
                                                 const float* __restrict__ wdw,
                                                 float* __restrict__ out)
{
  int e = blockIdx.x * 256 + threadIdx.x;       // 128*HW elements exactly
  int c = e / HW, p = e - c * HW;
  int h = p / Wd, w = p - h * Wd;
  const float* xc = xin + (size_t)c * HW;
  float acc = 0.f;
#pragma unroll
  for (int ky = 0; ky < 3; ++ky)
#pragma unroll
    for (int kx = 0; kx < 3; ++kx) {
      int hy = h + ky - 1, wx = w + kx - 1;
      if (hy >= 0 && hy < Hd && wx >= 0 && wx < Wd)
        acc += wdw[c * 9 + ky * 3 + kx] * xc[hy * Wd + wx];
    }
  out[e] = gelu_f(acc);
}

// DCN modulator: dwconv3 + bias -> LayerNorm(128) -> GELU; wave per pixel (wave32).
// Output is bf16 (only consumed by the offsets/mask/cfs GEMM).
__global__ __launch_bounds__(256) void dcn_pre_k(
    const float* __restrict__ x2, const float* __restrict__ wdw,
    const float* __restrict__ bdw, const float* __restrict__ lng,
    const float* __restrict__ lnb, bf16_t* __restrict__ x1i)
{
  const int p    = blockIdx.x * 8 + (threadIdx.x >> 5);
  const int lane = threadIdx.x & 31;
  const int h = p / Wd, w = p - h * Wd;
  float v[4], s1 = 0.f, s2 = 0.f;
#pragma unroll
  for (int q = 0; q < 4; ++q) {
    int c = lane + q * 32;
    const float* xc = x2 + (size_t)c * HW;
    float acc = bdw[c];
#pragma unroll
    for (int ky = 0; ky < 3; ++ky)
#pragma unroll
      for (int kx = 0; kx < 3; ++kx) {
        int hy = h + ky - 1, wx = w + kx - 1;
        if (hy >= 0 && hy < Hd && wx >= 0 && wx < Wd)
          acc += wdw[c * 9 + ky * 3 + kx] * xc[hy * Wd + wx];
      }
    v[q] = acc; s1 += acc; s2 += acc * acc;
  }
#pragma unroll
  for (int m = 16; m >= 1; m >>= 1) {
    s1 += __shfl_xor(s1, m, 32);
    s2 += __shfl_xor(s2, m, 32);
  }
  float mean = s1 * (1.f / 128.f);
  float var  = s2 * (1.f / 128.f) - mean * mean;
  float rstd = rsqrtf(var + 1e-5f);
#pragma unroll
  for (int q = 0; q < 4; ++q) {
    int c = lane + q * 32;
    float yv = lng[c] * (v[q] - mean) * rstd + lnb[c];
    x1i[(size_t)c * HW + p] = f2bf(gelu_f(yv));
  }
}

// float -> bf16 bulk conversion (inputs / weights, once per call)
__global__ void cvt_bf16_k(const float* __restrict__ src, bf16_t* __restrict__ dst,
                           long n)
{
  long i = (long)blockIdx.x * 256 + threadIdx.x;
  if (i < n) dst[i] = f2bf(src[i]);
}

// pack w_off|w_mask|cfs_w -> Wcat[112x128] bf16, biases -> bcat[112] f32
__global__ void pack_wcat(const float* __restrict__ w_off, const float* __restrict__ b_off,
                          const float* __restrict__ w_mask, const float* __restrict__ b_mask,
                          const float* __restrict__ cfs_w, const float* __restrict__ cfs_b,
                          bf16_t* __restrict__ wcat, float* __restrict__ bcat)
{
  int i = blockIdx.x * 256 + threadIdx.x;
  if (i < 112 * 128) {
    int r = i >> 7, k = i & 127;
    float v;
    if (r < 72)        v = w_off[r * 128 + k];
    else if (r < 108)  v = w_mask[(r - 72) * 128 + k];
    else               v = cfs_w[(r - 108) * 128 + k];
    wcat[i] = f2bf(v);
  }
  if (i < 112)
    bcat[i] = (i < 72) ? b_off[i] : (i < 108 ? b_mask[i - 72] : cfs_b[i - 108]);
}

__global__ void zero_k(float* __restrict__ p, long n)
{
  long i = (long)blockIdx.x * 256 + threadIdx.x;
  if (i < n) p[i] = 0.f;
}

// deformable sampling: wave per (pixel,group), lane = channel-in-group; bf16 out.
__global__ __launch_bounds__(256) void dcn_sample_k(
    const float* __restrict__ S, const float* __restrict__ xpad,
    bf16_t* __restrict__ dout)
{
  const int unit = blockIdx.x * 8 + (threadIdx.x >> 5);
  const int lane = threadIdx.x & 31;
  const int p = unit >> 2, g = unit & 3;
  const int h = p / Wd, w = p - h * Wd;

  float lg[9], mxv = -1e30f;
#pragma unroll
  for (int k = 0; k < 9; ++k) {
    lg[k] = S[(size_t)(72 + g * 9 + k) * HW + p];
    mxv = fmaxf(mxv, lg[k]);
  }
  float den = 0.f;
#pragma unroll
  for (int k = 0; k < 9; ++k) { lg[k] = __expf(lg[k] - mxv); den += lg[k]; }
  const float inv = 1.f / den;

  const int C = g * 32 + lane;
  const float* img = xpad + (size_t)C * (HI * HI);
  float acc = 0.f;
#pragma unroll
  for (int k = 0; k < 9; ++k) {
    float ox = S[(size_t)((g * 9 + k) * 2 + 0) * HW + p];
    float oy = S[(size_t)((g * 9 + k) * 2 + 1) * HW + p];
    float xs = (float)(w + 1) + (float)(k / 3) - 1.f + ox;
    float ys = (float)(h + 1) + (float)(k % 3) - 1.f + oy;
    float x0f = floorf(xs), y0f = floorf(ys);
    float wx = xs - x0f, wy = ys - y0f;
    int x0 = (int)x0f, y0 = (int)y0f;
    float v00 = (x0 >= 0 && x0 < HI && y0 >= 0 && y0 < HI)         ? img[y0 * HI + x0]           : 0.f;
    float v10 = (x0+1 >= 0 && x0+1 < HI && y0 >= 0 && y0 < HI)     ? img[y0 * HI + x0 + 1]       : 0.f;
    float v01 = (x0 >= 0 && x0 < HI && y0+1 >= 0 && y0+1 < HI)     ? img[(y0 + 1) * HI + x0]     : 0.f;
    float v11 = (x0+1 >= 0 && x0+1 < HI && y0+1 >= 0 && y0+1 < HI) ? img[(y0 + 1) * HI + x0 + 1] : 0.f;
    float bil = v00 * (1.f - wx) * (1.f - wy) + v10 * wx * (1.f - wy)
              + v01 * (1.f - wx) * wy         + v11 * wx * wy;
    acc += lg[k] * inv * bil;
  }
  float sig    = 1.f / (1.f + __expf(-S[(size_t)(108 + g) * HW + p]));
  float center = img[(h + 1) * HI + (w + 1)];
  dout[(size_t)C * HW + p] = f2bf(acc * (1.f - sig) + center * sig);
}

extern "C" void kernel_launch(void* const* d_in, const int* in_sizes, int n_in,
                              void* d_out, int out_size, void* d_ws, size_t ws_size,
                              hipStream_t stream)
{
  (void)in_sizes; (void)n_in; (void)out_size; (void)ws_size;
  const float* x      = (const float*)d_in[0];
  const float* w_in   = (const float*)d_in[1];
  const float* dctmix = (const float*)d_in[2];
  const float* w_dw   = (const float*)d_in[3];
  const float* w_out  = (const float*)d_in[4];
  const float* dcnw   = (const float*)d_in[5];
  const float* dcnb   = (const float*)d_in[6];
  const float* ln_g   = (const float*)d_in[7];
  const float* ln_b   = (const float*)d_in[8];
  const float* w_off  = (const float*)d_in[9];
  const float* b_off  = (const float*)d_in[10];
  const float* w_mask = (const float*)d_in[11];
  const float* b_mask = (const float*)d_in[12];
  const float* w_ip   = (const float*)d_in[13];
  const float* b_ip   = (const float*)d_in[14];
  const float* w_op   = (const float*)d_in[15];
  const float* b_op   = (const float*)d_in[16];
  const float* cfs_w  = (const float*)d_in[17];
  const float* cfs_b  = (const float*)d_in[18];
  float* out = (float*)d_out;

  // workspace carve-up (byte based, 256B aligned slices)
  char* w8 = (char*)d_ws;
  auto take = [&](size_t bytes) {
    char* p = w8; w8 += (bytes + 255) & ~(size_t)255; return p;
  };
  float*  y0    = (float*)take((size_t)256 * HW * 4);   // proj_in out, DCT in-place
  float*  x1g   = (float*)take((size_t)128 * HW * 4);   // gelu(dwconv(x1)) gate (f32 aux)
  float*  xpad  = (float*)take((size_t)128 * HI * HI * 4); // padded input_proj image
  float*  Sb    = (float*)take((size_t)112 * HW * 4);   // offsets|mask|cfs
  float*  bcat  = (float*)take(128 * 4);
  bf16_t* xb    = (bf16_t*)take((size_t)64  * HW * 2);  // input, bf16
  bf16_t* x2b   = (bf16_t*)take((size_t)128 * HW * 2);  // DCT x2 mirror, bf16
  bf16_t* x1ib  = (bf16_t*)take((size_t)128 * HW * 2);  // modulator, bf16
  bf16_t* dbufb = (bf16_t*)take((size_t)128 * HW * 2);  // sampled DCN feats, bf16
  bf16_t* hbufb = (bf16_t*)take((size_t)128 * HW * 2);  // gated hidden, bf16
  bf16_t* wbin  = (bf16_t*)take((size_t)256 * 64 * 2);
  bf16_t* wbip  = (bf16_t*)take((size_t)128 * 128 * 2);
  bf16_t* wbop  = (bf16_t*)take((size_t)128 * 128 * 2);
  bf16_t* wbout = (bf16_t*)take((size_t)64 * 128 * 2);
  bf16_t* wcat  = (bf16_t*)take((size_t)112 * 128 * 2);

  dim3 blk(256);
  long npad = (long)128 * HI * HI;
  zero_k<<<(unsigned)((npad + 255) / 256), blk, 0, stream>>>(xpad, npad);
  cvt_bf16_k<<<(64 * HW) / 256, blk, 0, stream>>>(x, xb, (long)64 * HW);
  cvt_bf16_k<<<64, blk, 0, stream>>>(w_in, wbin, 256 * 64);
  cvt_bf16_k<<<64, blk, 0, stream>>>(w_ip, wbip, 128 * 128);
  cvt_bf16_k<<<64, blk, 0, stream>>>(w_op, wbop, 128 * 128);
  cvt_bf16_k<<<32, blk, 0, stream>>>(w_out, wbout, 64 * 128);
  pack_wcat<<<56, blk, 0, stream>>>(w_off, b_off, w_mask, b_mask, cfs_w, cfs_b,
                                    wcat, bcat);

  gemm_wmma<256, 64, EPI_STORE><<<576, blk, 0, stream>>>(wbin, xb, y0, nullptr, nullptr);
  dct_k<<<576, blk, 0, stream>>>(y0, dctmix, x2b);
  dw_gelu_k<<<(128 * HW) / 256, blk, 0, stream>>>(y0, w_dw, x1g);
  dcn_pre_k<<<HW / 8, blk, 0, stream>>>(y0 + (size_t)128 * HW, dcnw, dcnb,
                                        ln_g, ln_b, x1ib);
  gemm_wmma<128, 128, EPI_PAD><<<576, blk, 0, stream>>>(wbip, x2b, xpad, b_ip, nullptr);
  gemm_wmma<112, 128, EPI_BIAS><<<576, blk, 0, stream>>>(wcat, x1ib, Sb, bcat, nullptr);
  dcn_sample_k<<<(HW * 4) / 8, blk, 0, stream>>>(Sb, xpad, dbufb);
  gemm_wmma<128, 128, EPI_GATE><<<576, blk, 0, stream>>>(wbop, dbufb, hbufb, b_op, x1g);
  gemm_wmma<64, 128, EPI_STORE><<<576, blk, 0, stream>>>(wbout, hbufb, out, nullptr, nullptr);
}
